// AdaFNO2D_52321291600564
// MI455X (gfx1250) — matrix operand order repaired and checked by reading, not compiled, not run on hardware
//
#include <hip/hip_runtime.h>
#include <hip/hip_bf16.h>

// ---------------------------------------------------------------------------
// AdaFNO via truncated DFT + WMMA (gfx1250, wave32)
//   mask keeps kx in {2..16} u {112..126} (30), ky in {2..16} (15) -> 450 bins
//   forward ortho 1/128 folded into Fx; inverse hermitian factor 2/128 = 1/64.
//   All LDS fragment traffic is vectorized (ds_load_b128 / ds_store_b128):
//     A fragments: row-major, ld % 8 == 0  -> 2x b128 per fragment
//     B fragments: transposed [N][K]       -> 2x b128 per fragment
//     D tiles    : flipped GEMMs so per-lane elements are contiguous -> b128
// ---------------------------------------------------------------------------

typedef __attribute__((ext_vector_type(16))) _Float16     v16h;
typedef __attribute__((ext_vector_type(8)))  _Float16     v8h;
typedef __attribute__((ext_vector_type(8)))  float        v8f;
typedef __attribute__((ext_vector_type(8)))  unsigned int v8ui;

#define TWO_PI_OVER_128 0.049087385212340526f
#define LN10000 9.210340371976184f
#define SHRINK_L 0.01f

// A 16x32 f16 fragment from row-major [M][K] (ld multiple of 8 halves)
__device__ __forceinline__ v16h frag_a(const _Float16* __restrict__ s, int ld,
                                       int row0, int k0, int lane) {
  const _Float16* p = s + (row0 + (lane & 15)) * ld + k0 + 8 * (lane >> 4);
  v8h lo = *(const v8h*)p;        // halves 8h .. 8h+7
  v8h hi = *(const v8h*)(p + 16); // halves 16+8h .. 16+8h+7
  return __builtin_shufflevector(lo, hi, 0, 1, 2, 3, 4, 5, 6, 7,
                                 8, 9, 10, 11, 12, 13, 14, 15);
}
// B 32x16 f16 fragment from TRANSPOSED storage [N][K] (ldT multiple of 8)
__device__ __forceinline__ v16h frag_bT(const _Float16* __restrict__ s, int ldT,
                                        int k0, int col0, int lane) {
  const _Float16* p = s + (col0 + (lane & 15)) * ldT + k0 + 8 * (lane >> 4);
  v8h lo = *(const v8h*)p;        // halves 8h .. 8h+7
  v8h hi = *(const v8h*)(p + 16); // halves 16+8h .. 16+8h+7
  return __builtin_shufflevector(lo, hi, 0, 1, 2, 3, 4, 5, 6, 7,
                                 8, 9, 10, 11, 12, 13, 14, 15);
}
__device__ __forceinline__ v16h fneg16(v16h a) {
  v8ui u = __builtin_bit_cast(v8ui, a);
  u = u ^ (v8ui)(0x80008000u);  // packed f16 sign flip
  return __builtin_bit_cast(v16h, u);
}
__device__ __forceinline__ v8f wmma16(v16h a, v16h b, v8f c) {
  return __builtin_amdgcn_wmma_f32_16x16x32_f16(false, a, false, b,
                                                (short)0, c, false, false);
}
__device__ __forceinline__ v8h pack8(v8f c) {
  v8h d;
#pragma unroll
  for (int e = 0; e < 8; ++e) d[e] = (_Float16)c[e];
  return d;
}
__device__ __forceinline__ int kx_of(int m) { return (m < 15) ? m + 2 : m + 97; }

// ===========================================================================
// Kernel 1: truncated forward DFT + alpha*FREQ_EMB.  1 block per (b,c).
//   Stage1 (flipped): Y1^T[y][m] = sum_x imgT[y][x] * FxT[x][m]
//     A = imgT row-major [y][x] (ld 136), B = Fx row-major [m][x] as [N][K]
//     D rows = y vary per lane -> b128 store into Y1 row-major [m][y] (ld 136)
//   Stage2: X[m][ky] = sum_y Y1[m][y] * Fy[y][ky]
//     A = Y1 [m][y] (ld 136), B = FyT [ky][y] (ld 136)
// ===========================================================================
__global__ void __launch_bounds__(256)
k_fwd(const float* __restrict__ x, const float* __restrict__ alpha,
      float* __restrict__ XMr, float* __restrict__ XMi) {
  __shared__ __align__(16) _Float16 sImgT[128 * 136];   // ~34 KB
  __shared__ __align__(16) _Float16 sFx[2 * 32 * 136];  // 17 KB, reused as Y1
  __shared__ __align__(16) _Float16 sFyT[2 * 16 * 136]; // 8.5 KB
  const int bc = blockIdx.x;
  const int c = bc % 768, b = bc / 768;
  const int tid = threadIdx.x, lane = tid & 31, wv = tid >> 5;
  const int h = lane >> 4, N = lane & 15;

  const float* __restrict__ src = x + (size_t)bc * (128 * 128);
  for (int i = tid; i < 128 * 128; i += 256) {
    __builtin_prefetch(src + ((i + 4096 < 16384) ? i + 4096 : i), 0, 1);
    int xx = i >> 7, yy = i & 127;
    sImgT[yy * 136 + xx] = (_Float16)src[i];
  }
  const float S = 1.0f / 128.0f;  // ortho forward
  for (int i = tid; i < 32 * 128; i += 256) {
    int m = i >> 7, xx = i & 127;
    float re = 0.f, im = 0.f;
    if (m < 30) {
      float ang = -TWO_PI_OVER_128 * (float)((kx_of(m) * xx) & 127);
      re = __cosf(ang) * S;
      im = __sinf(ang) * S;
    }
    sFx[m * 136 + xx] = (_Float16)re;
    sFx[32 * 136 + m * 136 + xx] = (_Float16)im;
  }
  for (int i = tid; i < 16 * 128; i += 256) {
    int n = i >> 7, yy = i & 127;
    float re = 0.f, im = 0.f;
    if (n < 15) {
      float ang = -TWO_PI_OVER_128 * (float)(((n + 2) * yy) & 127);
      re = __cosf(ang);
      im = __sinf(ang);
    }
    sFyT[n * 136 + yy] = (_Float16)re;
    sFyT[16 * 136 + n * 136 + yy] = (_Float16)im;
  }
  __syncthreads();

  // Stage 1: 2 parts x (8 y-tiles x 2 m-tiles) = 32 tiles; 4 per wave
  v8f acc[4];
#pragma unroll
  for (int i = 0; i < 4; ++i) {
    int t = wv * 4 + i;
    int part = t / 16, rest = t % 16, myt = rest / 2, nmt = rest % 2;
    v8f c0 = {};
    const _Float16* Bm = sFx + part * 32 * 136;
#pragma unroll
    for (int k0 = 0; k0 < 128; k0 += 32)
      c0 = wmma16(frag_a(sImgT, 136, myt * 16, k0, lane),
                  frag_bT(Bm, 136, k0, nmt * 16, lane), c0);
    acc[i] = c0;
  }
  __syncthreads();  // all Fx reads done before overwrite with Y1
#pragma unroll
  for (int i = 0; i < 4; ++i) {
    int t = wv * 4 + i;
    int part = t / 16, rest = t % 16, myt = rest / 2, nmt = rest % 2;
    _Float16* D = sFx + part * 32 * 136 + (nmt * 16 + N) * 136 + myt * 16 + 8 * h;
    *(v8h*)D = pack8(acc[i]);
  }
  __syncthreads();

  // Stage 2: X = Y1 * Fy (complex); 4 tiles, waves 0..3
  if (wv < 4) {
    const int p2 = wv & 1, mt = wv >> 1;
    const _Float16* Y1r = sFx;
    const _Float16* Y1i = sFx + 32 * 136;
    const _Float16* FyTr = sFyT;
    const _Float16* FyTi = sFyT + 16 * 136;
    v8f c0 = {};
#pragma unroll
    for (int k0 = 0; k0 < 128; k0 += 32) {
      if (p2 == 0) {  // Xr = Y1r*Fyr - Y1i*Fyi
        c0 = wmma16(frag_a(Y1r, 136, mt * 16, k0, lane),
                    frag_bT(FyTr, 136, k0, 0, lane), c0);
        c0 = wmma16(fneg16(frag_a(Y1i, 136, mt * 16, k0, lane)),
                    frag_bT(FyTi, 136, k0, 0, lane), c0);
      } else {        // Xi = Y1r*Fyi + Y1i*Fyr
        c0 = wmma16(frag_a(Y1r, 136, mt * 16, k0, lane),
                    frag_bT(FyTi, 136, k0, 0, lane), c0);
        c0 = wmma16(frag_a(Y1i, 136, mt * 16, k0, lane),
                    frag_bT(FyTr, 136, k0, 0, lane), c0);
      }
    }
    // alpha * FREQ_EMB (real part only) + store masked bins
    const int nb = c / 96, cc = c % 96;
    const float al = alpha[c];
    const bool useY = (c >= 384);
    const int j = (useY ? c - 384 : c) >> 1;
    const float dv = __expf(-LN10000 * (float)(2 * j) * (1.0f / 384.0f));
    const bool isCos = (c & 1);
    float* __restrict__ dst = (p2 == 0) ? XMr : XMi;
#pragma unroll
    for (int e = 0; e < 8; ++e) {
      int m = mt * 16 + e + 8 * h;
      if (m < 30 && N < 15) {
        float v = c0[e];
        if (p2 == 0) {
          float pos = useY ? (float)(N + 2) : (float)kx_of(m);
          float arg = pos * dv;
          v += al * (isCos ? __cosf(arg) : __sinf(arg));
        }
        int g = b * 450 + m * 15 + N;
        dst[((size_t)nb * 3600 + g) * 96 + cc] = v;
      }
    }
  }
}

// ===========================================================================
// Kernel 2: block-diagonal complex MLP + soft-shrink. Both layers flipped:
//   out^T[o][row] = W^T[o][k] . act^T[k][row]
//   A = W^T row-major [o][k] (ld 104); B = activations row-major [row][k]
//   D rows = o vary per lane -> b128 store into next layer's B layout /
//   coalesced float4 global stores.
// grid (113 slabs of 32 rows, 8 blocks); M = B*450 = 3600
// ===========================================================================
__global__ void __launch_bounds__(256)
k_mlp(const float* __restrict__ XMr, const float* __restrict__ XMi,
      const float* __restrict__ w1r, const float* __restrict__ b1r,
      const float* __restrict__ w1i, const float* __restrict__ b1i,
      const float* __restrict__ w2r, const float* __restrict__ b2r,
      const float* __restrict__ w2i, const float* __restrict__ b2i,
      float* __restrict__ YMr, float* __restrict__ YMi) {
  __shared__ __align__(16) _Float16 sX[2 * 32 * 104];  // xr,xi -> r1,i1  13 KB
  __shared__ __align__(16) _Float16 sW[2 * 96 * 104];  // Wa^T,Wb^T       40 KB
  __shared__ float sBias[4 * 96];
  const int slab = blockIdx.x, n = blockIdx.y;
  const int tid = threadIdx.x, lane = tid & 31, wv = tid >> 5;
  const int r0 = slab * 32;
  const int h = lane >> 4, N = lane & 15;

  for (int i = tid; i < 32 * 96; i += 256) {
    int rr = i / 96, kk = i % 96, row = r0 + rr;
    float vr = 0.f, vi = 0.f;
    if (row < 3600) {
      size_t off = ((size_t)n * 3600 + row) * 96 + kk;
      vr = XMr[off];
      vi = XMi[off];
    }
    sX[rr * 104 + kk] = (_Float16)vr;
    sX[32 * 104 + rr * 104 + kk] = (_Float16)vi;
  }
  for (int i = tid; i < 96 * 96; i += 256) {  // transpose [k][o] -> [o][k]
    int kk = i / 96, oo = i % 96;
    sW[oo * 104 + kk] = (_Float16)w1r[(size_t)n * 9216 + i];
    sW[96 * 104 + oo * 104 + kk] = (_Float16)w1i[(size_t)n * 9216 + i];
  }
  for (int i = tid; i < 96; i += 256) {
    sBias[i] = b1r[(size_t)n * 96 + i];
    sBias[96 + i] = b1i[(size_t)n * 96 + i];
    sBias[192 + i] = b2r[(size_t)n * 96 + i];
    sBias[288 + i] = b2i[(size_t)n * 96 + i];
  }
  __syncthreads();

  // Layer 1: 24 tiles (part, 6 o-tiles, 2 row-tiles); 3 per wave
  v8h st[3];
#pragma unroll
  for (int i = 0; i < 3; ++i) {
    int t = wv * 3 + i;
    int part = t / 12, rest = t % 12, ot = rest % 6, rt = rest / 6;
    v8f c0 = {};
#pragma unroll
    for (int k0 = 0; k0 < 96; k0 += 32) {
      if (part == 0) {  // r1^T = w1r^T xr^T - w1i^T xi^T
        c0 = wmma16(frag_a(sW, 104, ot * 16, k0, lane),
                    frag_bT(sX, 104, k0, rt * 16, lane), c0);
        c0 = wmma16(fneg16(frag_a(sW + 96 * 104, 104, ot * 16, k0, lane)),
                    frag_bT(sX + 32 * 104, 104, k0, rt * 16, lane), c0);
      } else {          // i1^T = w1r^T xi^T + w1i^T xr^T
        c0 = wmma16(frag_a(sW, 104, ot * 16, k0, lane),
                    frag_bT(sX + 32 * 104, 104, k0, rt * 16, lane), c0);
        c0 = wmma16(frag_a(sW + 96 * 104, 104, ot * 16, k0, lane),
                    frag_bT(sX, 104, k0, rt * 16, lane), c0);
      }
    }
    const float* bi = sBias + part * 96;
#pragma unroll
    for (int e = 0; e < 8; ++e) c0[e] += bi[ot * 16 + 8 * h + e];
    st[i] = pack8(c0);
  }
  __syncthreads();  // all X reads done before overwrite with r1/i1
#pragma unroll
  for (int i = 0; i < 3; ++i) {
    int t = wv * 3 + i;
    int part = t / 12, rest = t % 12, ot = rest % 6, rt = rest / 6;
    *(v8h*)(sX + part * 32 * 104 + (rt * 16 + N) * 104 + ot * 16 + 8 * h) = st[i];
  }
  for (int i = tid; i < 96 * 96; i += 256) {
    int kk = i / 96, oo = i % 96;
    sW[oo * 104 + kk] = (_Float16)w2r[(size_t)n * 9216 + i];
    sW[96 * 104 + oo * 104 + kk] = (_Float16)w2i[(size_t)n * 9216 + i];
  }
  __syncthreads();

  // Layer 2 + shrink + coalesced store
#pragma unroll
  for (int i = 0; i < 3; ++i) {
    int t = wv * 3 + i;
    int part = t / 12, rest = t % 12, ot = rest % 6, rt = rest / 6;
    v8f c0 = {};
#pragma unroll
    for (int k0 = 0; k0 < 96; k0 += 32) {
      if (part == 0) {  // r2^T = w2r^T r1^T - w2i^T i1^T
        c0 = wmma16(frag_a(sW, 104, ot * 16, k0, lane),
                    frag_bT(sX, 104, k0, rt * 16, lane), c0);
        c0 = wmma16(fneg16(frag_a(sW + 96 * 104, 104, ot * 16, k0, lane)),
                    frag_bT(sX + 32 * 104, 104, k0, rt * 16, lane), c0);
      } else {          // i2^T = w2r^T i1^T + w2i^T r1^T
        c0 = wmma16(frag_a(sW, 104, ot * 16, k0, lane),
                    frag_bT(sX + 32 * 104, 104, k0, rt * 16, lane), c0);
        c0 = wmma16(frag_a(sW + 96 * 104, 104, ot * 16, k0, lane),
                    frag_bT(sX, 104, k0, rt * 16, lane), c0);
      }
    }
    int row = r0 + rt * 16 + N;
    if (row < 3600) {
      const float* bi = sBias + 192 + part * 96;
      float out8[8];
#pragma unroll
      for (int e = 0; e < 8; ++e) {
        float v = c0[e] + bi[ot * 16 + 8 * h + e];
        float av = fabsf(v) - SHRINK_L;
        out8[e] = (av > 0.f) ? copysignf(av, v) : 0.f;
      }
      float* __restrict__ dst = ((part == 0) ? YMr : YMi) +
                                ((size_t)n * 3600 + row) * 96 + ot * 16 + 8 * h;
      *(float4*)dst = make_float4(out8[0], out8[1], out8[2], out8[3]);
      *(float4*)(dst + 4) = make_float4(out8[4], out8[5], out8[6], out8[7]);
    }
  }
}

// ===========================================================================
// Kernel 3: truncated inverse DFT. 1 block per (b,c).
//   Stage A: Z[m][y] = X[m][ky] * Ey[ky][y]   (A = X [32][40], B = EyT [y][ky])
//     D cols = y fixed per lane -> b128 store into ZT[y][m] (ld 40)
//   Stage B (flipped): out^T[y][x] = ZT[y][k] . Ex^T[k][x]
//     A = ZT row-major [y][k] (ld 40), B = Ex row-major [x][k] as [N][K]
//     D rows = y vary per lane -> two coalesced float4 global stores
// ===========================================================================
__global__ void __launch_bounds__(256)
k_inv(const float* __restrict__ YMr, const float* __restrict__ YMi,
      float* __restrict__ out) {
  __shared__ __align__(16) _Float16 sU[2 * 128 * 40];    // X (2x32x40) then Ex
  __shared__ __align__(16) _Float16 sEyT[2 * 128 * 40];  // 20 KB
  __shared__ __align__(16) _Float16 sZT[2 * 128 * 40];   // 20 KB
  const int bc = blockIdx.x;
  const int c = bc % 768, b = bc / 768;
  const int nb = c / 96, cc = c % 96;
  const int tid = threadIdx.x, lane = tid & 31, wv = tid >> 5;
  const int h = lane >> 4, N = lane & 15;

  for (int i = tid; i < 32 * 32; i += 256) {
    int m = i >> 5, k = i & 31;
    float vr = 0.f, vi = 0.f;
    if (m < 30 && k < 15) {
      size_t off = ((size_t)nb * 3600 + b * 450 + m * 15 + k) * 96 + cc;
      vr = YMr[off];
      vi = YMi[off];
    }
    sU[m * 40 + k] = (_Float16)vr;
    sU[32 * 40 + m * 40 + k] = (_Float16)vi;
  }
  for (int i = tid; i < 128 * 32; i += 256) {  // EyT[y][ky]
    int y = i >> 5, k = i & 31;
    float re = 0.f, im = 0.f;
    if (k < 15) {
      float ang = TWO_PI_OVER_128 * (float)(((k + 2) * y) & 127);
      re = __cosf(ang);
      im = __sinf(ang);
    }
    sEyT[y * 40 + k] = (_Float16)re;
    sEyT[128 * 40 + y * 40 + k] = (_Float16)im;
  }
  __syncthreads();

  // Stage A: 2 parts x (2 m-tiles x 8 y-tiles) = 32 tiles; 4 per wave
#pragma unroll
  for (int i = 0; i < 4; ++i) {
    int t = wv * 4 + i;
    int part = t / 16, rest = t % 16, mt = rest / 8, nt = rest % 8;
    v8f c0 = {};
    if (part == 0) {  // Zr = Xr*Eyr - Xi*Eyi
      c0 = wmma16(frag_a(sU, 40, mt * 16, 0, lane),
                  frag_bT(sEyT, 40, 0, nt * 16, lane), c0);
      c0 = wmma16(fneg16(frag_a(sU + 32 * 40, 40, mt * 16, 0, lane)),
                  frag_bT(sEyT + 128 * 40, 40, 0, nt * 16, lane), c0);
    } else {          // Zi = Xr*Eyi + Xi*Eyr
      c0 = wmma16(frag_a(sU, 40, mt * 16, 0, lane),
                  frag_bT(sEyT + 128 * 40, 40, 0, nt * 16, lane), c0);
      c0 = wmma16(frag_a(sU + 32 * 40, 40, mt * 16, 0, lane),
                  frag_bT(sEyT, 40, 0, nt * 16, lane), c0);
    }
    *(v8h*)(sZT + part * 128 * 40 + (nt * 16 + N) * 40 + mt * 16 + 8 * h) =
        pack8(c0);
  }
  __syncthreads();  // X reads done; overwrite sU with Ex
  for (int i = tid; i < 128 * 32; i += 256) {  // Ex[x][kx]
    int xx = i >> 5, k = i & 31;
    float re = 0.f, im = 0.f;
    if (k < 30) {
      float ang = TWO_PI_OVER_128 * (float)((kx_of(k) * xx) & 127);
      re = __cosf(ang);
      im = __sinf(ang);
    }
    sU[xx * 40 + k] = (_Float16)re;
    sU[128 * 40 + xx * 40 + k] = (_Float16)im;
  }
  __syncthreads();

  // Stage B: out^T; y-tile = wave, 8 x-tiles each; single K=32 step
  float* __restrict__ dst = out + (size_t)bc * (128 * 128);
#pragma unroll
  for (int nt = 0; nt < 8; ++nt) {
    v8f c0 = {};
    c0 = wmma16(frag_a(sZT, 40, wv * 16, 0, lane),
                frag_bT(sU, 40, 0, nt * 16, lane), c0);
    c0 = wmma16(fneg16(frag_a(sZT + 128 * 40, 40, wv * 16, 0, lane)),
                frag_bT(sU + 128 * 40, 40, 0, nt * 16, lane), c0);
    int xx = nt * 16 + N;
    float* p = dst + xx * 128 + wv * 16 + 8 * h;
    *(float4*)p = make_float4(c0[0] * (1.0f / 64.0f), c0[1] * (1.0f / 64.0f),
                              c0[2] * (1.0f / 64.0f), c0[3] * (1.0f / 64.0f));
    *(float4*)(p + 4) = make_float4(c0[4] * (1.0f / 64.0f), c0[5] * (1.0f / 64.0f),
                                    c0[6] * (1.0f / 64.0f), c0[7] * (1.0f / 64.0f));
  }
}

// ===========================================================================
extern "C" void kernel_launch(void* const* d_in, const int* in_sizes, int n_in,
                              void* d_out, int out_size, void* d_ws, size_t ws_size,
                              hipStream_t stream) {
  const float* x     = (const float*)d_in[0];
  const float* alpha = (const float*)d_in[1];
  const float* w1r   = (const float*)d_in[2];
  const float* b1r   = (const float*)d_in[3];
  const float* w1i   = (const float*)d_in[4];
  const float* b1i   = (const float*)d_in[5];
  const float* w2r   = (const float*)d_in[6];
  const float* b2r   = (const float*)d_in[7];
  const float* w2i   = (const float*)d_in[8];
  const float* b2i   = (const float*)d_in[9];
  float* out = (float*)d_out;

  const size_t NSP = (size_t)8 * 3600 * 96;  // per spectral plane
  float* XMr = (float*)d_ws;
  float* XMi = XMr + NSP;
  float* YMr = XMi + NSP;
  float* YMi = YMr + NSP;

  k_fwd<<<dim3(8 * 768), dim3(256), 0, stream>>>(x, alpha, XMr, XMi);
  k_mlp<<<dim3(113, 8), dim3(256), 0, stream>>>(XMr, XMi, w1r, b1r, w1i, b1i,
                                                w2r, b2r, w2i, b2i, YMr, YMi);
  k_inv<<<dim3(8 * 768), dim3(256), 0, stream>>>(YMr, YMi, out);
}
